// HeadEfficient_85761906967091
// MI455X (gfx1250) — compile-verified, matrix operand order checked by reading, not run on hardware
//
#include <hip/hip_runtime.h>
#include <hip/hip_bf16.h>

// B=8, T=4096, EMB=1024, HEAD=64.  out = softmax(mask(QK^T/8)) V, f32 out.
// Phase 0: W[1024,192] f32 -> Wt[192,1024] bf16 (L2-resident).
// Phase 1: bandwidth-bound QKV projection (x streams at HBM rate, 12 WMMAs per
//          A-fragment), writes K[T,64], Q[T,64], V^T[64,T] bf16 (12 MB -> L2).
// Phase 2: flash attention; K/V tiles staged once per workgroup into LDS via
//          global_load_async_to_lds_b128 (double-buffered, ASYNCcnt-tracked),
//          WMMA reads fragments from LDS while the next tile streams in.

typedef __attribute__((ext_vector_type(16))) __bf16 v16bf;
typedef __attribute__((ext_vector_type(8)))  __bf16 v8bf;
typedef __attribute__((ext_vector_type(8)))  float  v8f;

#define B_    8
#define T_    4096
#define EMB_  1024
#define HEAD_ 64
#define NKQV  192

// Load a 16-bit A/B WMMA fragment: elements 0..7 from base+o1, 8..15 from
// base+o2 (o1 = half*8, o2 = 16+half*8 per the CDNA5 16-bit 16x32 layout).
static __device__ __forceinline__ v16bf load_frag(const __bf16* base, int o1, int o2) {
  v8bf lo = *(const v8bf*)(base + o1);
  v8bf hi = *(const v8bf*)(base + o2);
  v16bf r;
#pragma unroll
  for (int i = 0; i < 8; ++i) { r[i] = lo[i]; r[i + 8] = hi[i]; }
  return r;
}

// Low 32 bits of a generic pointer into LDS = byte offset within the wave's
// LDS allocation (flat LDS aperture truncates the upper bits).
static __device__ __forceinline__ unsigned lds_off(const void* p) {
  return (unsigned)(size_t)p;
}

// One lane moves 16B global -> LDS, tracked by ASYNCcnt (CDNA5 async copy).
static __device__ __forceinline__ void async_cp16(unsigned dst_lds, const void* src) {
  asm volatile("global_load_async_to_lds_b128 %0, %1, off"
               :: "v"(dst_lds), "v"((unsigned long long)(size_t)src)
               : "memory");
}

// ---------------- Phase 0: transpose+convert W ----------------
__global__ void wt_convert_kernel(const float* __restrict__ W, __bf16* __restrict__ Wt) {
  int idx = blockIdx.x * blockDim.x + threadIdx.x;
  if (idx >= NKQV * EMB_) return;
  int n = idx >> 10;        // output col of W (0..191)
  int k = idx & (EMB_ - 1); // K index
  Wt[(size_t)n * EMB_ + k] = (__bf16)W[(size_t)k * NKQV + n];
}

// ---------------- Phase 1: QKV projection ----------------
__global__ __launch_bounds__(256)
void qkv_proj_kernel(const float* __restrict__ x, const __bf16* __restrict__ Wt,
                     __bf16* __restrict__ kbuf, __bf16* __restrict__ qbuf,
                     __bf16* __restrict__ vT) {
  const int lane = threadIdx.x & 31;
  const int half = lane >> 4;
  const int lr   = lane & 15;
  const int wave = blockIdx.x * (blockDim.x >> 5) + (threadIdx.x >> 5);
  const int m0   = wave * 16;                    // 16-row M tile of [B*T, EMB]

  v8f acc[12] = {};
  const float* xrow = x + (size_t)(m0 + lr) * EMB_;

  for (int kb = 0; kb < EMB_; kb += 32) {
    float4 a0 = *(const float4*)(xrow + kb + half * 8);
    float4 a1 = *(const float4*)(xrow + kb + half * 8 + 4);
    float4 a2 = *(const float4*)(xrow + kb + 16 + half * 8);
    float4 a3 = *(const float4*)(xrow + kb + 16 + half * 8 + 4);
    v16bf a;
    a[0]=(__bf16)a0.x; a[1]=(__bf16)a0.y; a[2]=(__bf16)a0.z; a[3]=(__bf16)a0.w;
    a[4]=(__bf16)a1.x; a[5]=(__bf16)a1.y; a[6]=(__bf16)a1.z; a[7]=(__bf16)a1.w;
    a[8]=(__bf16)a2.x; a[9]=(__bf16)a2.y; a[10]=(__bf16)a2.z; a[11]=(__bf16)a2.w;
    a[12]=(__bf16)a3.x; a[13]=(__bf16)a3.y; a[14]=(__bf16)a3.z; a[15]=(__bf16)a3.w;

#pragma unroll
    for (int n = 0; n < 12; ++n) {
      const __bf16* wrow = Wt + (size_t)(n * 16 + lr) * EMB_ + kb;
      v16bf bfr = load_frag(wrow, half * 8, 16 + half * 8);
      acc[n] = __builtin_amdgcn_wmma_f32_16x16x32_bf16(
          false, a, false, bfr, (short)0, acc[n], false, false);
    }
  }

  // Epilogue: C layout -> split into K / Q / V^T bf16 buffers.
#pragma unroll
  for (int n = 0; n < 12; ++n) {
    int col = n * 16 + lr;
#pragma unroll
    for (int r = 0; r < 8; ++r) {
      int row = m0 + half * 8 + r;               // global row in [0, B*T)
      __bf16 v = (__bf16)acc[n][r];
      if (col < 64) {
        kbuf[(size_t)row * HEAD_ + col] = v;
      } else if (col < 128) {
        qbuf[(size_t)row * HEAD_ + (col - 64)] = v;
      } else {
        int bi = row >> 12, t = row & (T_ - 1);
        vT[((size_t)bi * HEAD_ + (col - 128)) * T_ + t] = v;
      }
    }
  }
}

// ---------------- Phase 2: flash attention ----------------
__global__ __launch_bounds__(256)
void attn_kernel(const __bf16* __restrict__ kbuf, const __bf16* __restrict__ qbuf,
                 const __bf16* __restrict__ vT, float* __restrict__ out) {
  const int tid   = threadIdx.x;
  const int lane  = tid & 31;
  const int half  = lane >> 4;
  const int lr    = lane & 15;
  const int w     = tid >> 5;
  const int b     = blockIdx.x >> 5;
  const int qt    = blockIdx.x & 31;
  const int qbase = qt * 128 + w * 16;           // this wave's 16 query rows

  // Double-buffered shared K/V tiles (64 keys x 64 feats each, bf16) filled by
  // the async copy engine once per workgroup; wave-private P staging tile.
  __shared__ __align__(16) __bf16 Kt[2][64][64];     // [buf][key][feat]  2x8KB
  __shared__ __align__(16) __bf16 Vt[2][64][64];     // [buf][feat][key]  2x8KB
  __shared__ __align__(16) __bf16 pstage[8][16][72]; // 18KB

  // Issue the 4 async b128 copies this thread owns for key-tile j into buf.
  auto issue_tile = [&](int j, int buf) {
    const int k0 = j * 64;
    // K tile: contiguous 8KB block, thread covers bytes [tid*32, tid*32+32)
    const char* ksrc = (const char*)(kbuf + ((size_t)b * T_ + k0) * HEAD_);
    unsigned kdst = lds_off(&Kt[buf][0][0]) + tid * 32;
    async_cp16(kdst,      ksrc + tid * 32);
    async_cp16(kdst + 16, ksrc + tid * 32 + 16);
    // V^T tile: 64 rows (features) of 128B; thread -> (feat, 32B chunk)
    const int f = tid >> 2, ch = tid & 3;
    const char* vsrc = (const char*)(vT + ((size_t)b * HEAD_ + f) * T_ + k0);
    unsigned vdst = lds_off(&Vt[buf][f][0]) + ch * 32;
    async_cp16(vdst,      vsrc + ch * 32);
    async_cp16(vdst + 16, vsrc + ch * 32 + 16);
  };

  // Q fragments: A-matrix, K = feature dim (0..31, 32..63); resident all loop.
  v16bf aq0, aq1;
  {
    const __bf16* qrow = qbuf + ((size_t)b * T_ + qbase + lr) * HEAD_;
    aq0 = load_frag(qrow,      half * 8, 16 + half * 8);
    aq1 = load_frag(qrow + 32, half * 8, 16 + half * 8);
  }

  v8f o[4] = {};
  float mrun[8], lrun[8];
#pragma unroll
  for (int r = 0; r < 8; ++r) { mrun[r] = -__builtin_inff(); lrun[r] = 0.f; }

  const int jendw   = (qbase + 15) >> 6;   // last tile this wave computes
  const int jendblk = qt * 2 + 1;          // last tile the block stages

  issue_tile(0, 0);                        // preload

  for (int j = 0; j <= jendblk; ++j) {
    const int buf = j & 1;
    // Own async fills done, then barrier publishes every wave's fills.
    asm volatile("s_wait_asynccnt 0x0" ::: "memory");
    __syncthreads();
    // Stream next tile into the other buffer while we compute this one
    // (buffer (j+1)&1 was last read at iteration j-1, retired by the barrier).
    if (j < jendblk) issue_tile(j + 1, buf ^ 1);

    if (j > jendw) continue;               // fully-masked for this wave
    const int k0 = j * 64;

    // Scores S = Q K^T for 64 keys (4 N-subtiles x 2 K-chunks) from LDS
    v8f S[4];
#pragma unroll
    for (int nt = 0; nt < 4; ++nt) {
      const __bf16* krow = &Kt[buf][nt * 16 + lr][0];
      v16bf bk0 = load_frag(krow,      half * 8, 16 + half * 8);
      v16bf bk1 = load_frag(krow + 32, half * 8, 16 + half * 8);
      v8f s = {};
      s = __builtin_amdgcn_wmma_f32_16x16x32_bf16(false, aq0, false, bk0, (short)0, s, false, false);
      s = __builtin_amdgcn_wmma_f32_16x16x32_bf16(false, aq1, false, bk1, (short)0, s, false, false);
      S[nt] = s;
    }

    // scale by 1/sqrt(64), causal mask, new running max
    float mnew[8];
#pragma unroll
    for (int r = 0; r < 8; ++r) mnew[r] = mrun[r];
#pragma unroll
    for (int nt = 0; nt < 4; ++nt) {
      int key = k0 + nt * 16 + lr;
#pragma unroll
      for (int r = 0; r < 8; ++r) {
        int qrow = qbase + half * 8 + r;
        float val = S[nt][r] * 0.125f;
        if (key > qrow) val = -__builtin_inff();
        S[nt][r] = val;
        mnew[r] = fmaxf(mnew[r], val);
      }
    }
#pragma unroll
    for (int r = 0; r < 8; ++r)
#pragma unroll
      for (int mask = 1; mask < 16; mask <<= 1)
        mnew[r] = fmaxf(mnew[r], __shfl_xor(mnew[r], mask, 32));

    // rescale running state
    float rs[8];
#pragma unroll
    for (int r = 0; r < 8; ++r) {
      float alpha = __expf(mrun[r] - mnew[r]);
      lrun[r] *= alpha;
      rs[r] = 0.f;
#pragma unroll
      for (int nt = 0; nt < 4; ++nt) o[nt][r] *= alpha;
      mrun[r] = mnew[r];
    }

    // P = exp(S - m), row-sum, stage to LDS (C layout -> A layout transit)
#pragma unroll
    for (int nt = 0; nt < 4; ++nt) {
      int col = nt * 16 + lr;
#pragma unroll
      for (int r = 0; r < 8; ++r) {
        float p = __expf(S[nt][r] - mnew[r]);
        rs[r] += p;
        pstage[w][half * 8 + r][col] = (__bf16)p;
      }
    }
#pragma unroll
    for (int r = 0; r < 8; ++r) {
#pragma unroll
      for (int mask = 1; mask < 16; mask <<= 1)
        rs[r] += __shfl_xor(rs[r], mask, 32);
      lrun[r] += rs[r];
    }

    // DS ops from one wave complete in order; fence the compiler + counters.
    asm volatile("s_wait_dscnt 0" ::: "memory");

    // P as A fragments (K = key dim 0..31, 32..63)
    const __bf16* prow = &pstage[w][lr][0];
    v16bf ap0 = load_frag(prow,      half * 8, 16 + half * 8);
    v16bf ap1 = load_frag(prow + 32, half * 8, 16 + half * 8);

    // O += P V from the shared V^T tile
#pragma unroll
    for (int ntf = 0; ntf < 4; ++ntf) {
      const __bf16* vrow = &Vt[buf][ntf * 16 + lr][0];
      v16bf bv0 = load_frag(vrow,      half * 8, 16 + half * 8);
      v16bf bv1 = load_frag(vrow + 32, half * 8, 16 + half * 8);
      o[ntf] = __builtin_amdgcn_wmma_f32_16x16x32_bf16(false, ap0, false, bv0, (short)0, o[ntf], false, false);
      o[ntf] = __builtin_amdgcn_wmma_f32_16x16x32_bf16(false, ap1, false, bv1, (short)0, o[ntf], false, false);
    }
  }

  // normalize and store f32 output
#pragma unroll
  for (int ntf = 0; ntf < 4; ++ntf) {
    int col = ntf * 16 + lr;
#pragma unroll
    for (int r = 0; r < 8; ++r) {
      float inv = 1.f / lrun[r];
      out[((size_t)b * T_ + qbase + half * 8 + r) * HEAD_ + col] = o[ntf][r] * inv;
    }
  }
}

extern "C" void kernel_launch(void* const* d_in, const int* in_sizes, int n_in,
                              void* d_out, int out_size, void* d_ws, size_t ws_size,
                              hipStream_t stream) {
  (void)in_sizes; (void)n_in; (void)out_size; (void)ws_size;
  const float* x = (const float*)d_in[0];   // [8,4096,1024] f32
  const float* W = (const float*)d_in[1];   // [1024,192]    f32
  float* out = (float*)d_out;               // [8,4096,64]   f32

  char* ws = (char*)d_ws;
  __bf16* kbuf = (__bf16*)(ws);                                   // 4 MB
  __bf16* qbuf = (__bf16*)(ws + (size_t)4 * 1024 * 1024);         // 4 MB
  __bf16* vT   = (__bf16*)(ws + (size_t)8 * 1024 * 1024);         // 4 MB
  __bf16* Wt   = (__bf16*)(ws + (size_t)12 * 1024 * 1024);        // 384 KB

  wt_convert_kernel<<<(NKQV * EMB_ + 255) / 256, 256, 0, stream>>>(W, Wt);
  qkv_proj_kernel<<<(B_ * T_ / 16) / 8, 256, 0, stream>>>(x, Wt, kbuf, qbuf, vT);
  attn_kernel<<<B_ * (T_ / 128), 256, 0, stream>>>(kbuf, qbuf, vT, out);
}
